// DSQGAttentionN_11622181503116
// MI455X (gfx1250) — compile-verified, matrix operand order checked by reading, not run on hardware
//
#include <hip/hip_runtime.h>
#include <hip/hip_bf16.h>
#include <cmath>

// ---------------------------------------------------------------------------
// MI455X (gfx1250) implementation.
//  * x / weights converted once to f16 (TDM is a byte-mover; also halves GEMM
//    global traffic -- whole working set is L2-resident at 192MB anyway).
//  * GEMM tiles are DMA'd global->LDS by the Tensor Data Mover
//    (tensor_load_to_lds, TENSORcnt, double-buffered against the WMMA pipe),
//    with descriptor padding producing an 80B LDS row stride so fragment
//    ds_load_b128s are bank-conflict free.
//  * Matmul on v_wmma_f32_16x16x32_f16; all 4 B fragments loaded before the
//    4 back-to-back WMMAs.
//  * Attention (74 sparse offsets, HD=64, ~0.6 GFLOP) is VALU online-softmax.
// ---------------------------------------------------------------------------

typedef __attribute__((ext_vector_type(16))) _Float16 v16h;
typedef __attribute__((ext_vector_type(8)))  _Float16 v8h;
typedef __attribute__((ext_vector_type(4)))  _Float16 v4h;
typedef __attribute__((ext_vector_type(8)))  float    v8f;
typedef __attribute__((ext_vector_type(4)))  unsigned int v4u;
typedef __attribute__((ext_vector_type(8)))  int      v8i;
typedef __attribute__((ext_vector_type(4)))  int      v4i;

#define TILE_M 128
#define TILE_N 64
#define TILE_K 32
#define LDS_STRIDE 40   // halves per row; 80B stride (64B data + 16B TDM pad)

// Issue one TDM 2D tile load: rows x 32 f16 elements, row stride = K elems,
// destination LDS rows padded to 80B. Descriptor per CDNA5 ISA §8.3/8.4.
__device__ __forceinline__ void tdm_load_tile(const _Float16* gsrc, unsigned lds_byte_addr,
                                              int rows, int row_stride_elems)
{
    unsigned long long ga = (unsigned long long)(size_t)gsrc;
    v4u g0;
    g0.x = 1u;                                            // count=1 (valid user D#)
    g0.y = lds_byte_addr;                                 // lds_addr
    g0.z = (unsigned)(ga & 0xffffffffu);                  // global_addr[31:0]
    g0.w = (unsigned)((ga >> 32) & 0x01ffffffu) | (2u << 30); // addr[56:32] | type=2

    // group1: data_size=1(2B), pad_enable, pad_interval=3(16 DW=64B),
    //         pad_amount=3(4 DW=16B)
    unsigned w0 = (1u << 16) | (1u << 20) | (3u << 22) | (3u << 25);
    unsigned w1 = (unsigned)TILE_K << 16;                 // tensor_dim0=32 (bits 79:48)
    unsigned w2 = ((unsigned)rows & 0xffffu) << 16;       // tensor_dim1 (bits 111:80)
    unsigned w3 = (unsigned)TILE_K << 16;                 // tile_dim0=32 (bits 127:112)
    unsigned w4 = (unsigned)rows & 0xffffu;               // tile_dim1; tile_dim2=0
    unsigned w5 = (unsigned)row_stride_elems;             // tensor_dim0_stride[31:0]
    unsigned w6 = 0;                                      // stride hi / dim1_stride lo
    unsigned w7 = 0;
    union { unsigned u[8]; v8i v; } g1 = {{ w0, w1, w2, w3, w4, w5, w6, w7 }};
    v4i z4 = {};
    v8i z8 = {};
    // amdgpu-toolchain (clang-23) 6-arg form: (g0, g1, g2, g3, <v8i>, cpol)
    __builtin_amdgcn_tensor_load_to_lds(g0, g1.v, z4, z4, z8, 0);
}

// C[M,N] = A[M,K](f16) @ W[N,K](f16)^T + bias
//   epilogue 0: Cf (f32) = acc + bias
//   epilogue 1: Ch (f16) = mul * sigmoid(acc + bias)
__global__ __launch_bounds__(256)
void gemm_wmma_f16(const _Float16* __restrict__ A, const _Float16* __restrict__ W,
                   const float* __restrict__ bias, const float* __restrict__ mul,
                   float* __restrict__ Cf, _Float16* __restrict__ Ch,
                   int M, int N, int K, int epilogue)
{
    __shared__ __align__(16) _Float16 lA[2][TILE_M * LDS_STRIDE];
    __shared__ __align__(16) _Float16 lB[2][TILE_N * LDS_STRIDE];

    const int tid    = threadIdx.x;
    const int lane   = tid & 31;
    const int wave   = tid >> 5;               // 0..7
    const int blockM = blockIdx.y * TILE_M;
    const int blockN = blockIdx.x * TILE_N;

    const _Float16* Abase = A + (size_t)blockM * K;
    const _Float16* Wbase = W + (size_t)blockN * K;

    v8f acc[4] = {};

    // fragment indices per ISA 7.12.2 (16-bit A/B layout):
    // lanes 0-15 hold K {0..7,16..23}, lanes 16-31 hold K {8..15,24..31}
    const int mfrag = 16 * wave + (lane & 15);
    const int kb    = 8 * (lane >> 4);

    // prologue: DMA tile 0 into buffer 0
    if (wave == 0) {
        tdm_load_tile(Abase, (unsigned)(size_t)&lA[0][0], TILE_M, K);
        tdm_load_tile(Wbase, (unsigned)(size_t)&lB[0][0], TILE_N, K);
    }

    for (int k0 = 0; k0 < K; k0 += TILE_K) {
        const int cur = (k0 >> 5) & 1;
        const int nxt = cur ^ 1;

        __syncthreads();   // all waves done reading buffer 'nxt' (iter k-1)
        if (wave == 0) {
            if (k0 + TILE_K < K) {
                tdm_load_tile(Abase + k0 + TILE_K, (unsigned)(size_t)&lA[nxt][0], TILE_M, K);
                tdm_load_tile(Wbase + k0 + TILE_K, (unsigned)(size_t)&lB[nxt][0], TILE_N, K);
                __builtin_amdgcn_s_wait_tensorcnt(2);  // in-order: tile k done
            } else {
                __builtin_amdgcn_s_wait_tensorcnt(0);
            }
        }
        __syncthreads();   // tile k visible to all waves

        union { v16h v; v8h h[2]; } afr;
        afr.h[0] = *(const v8h*)(&lA[cur][mfrag * LDS_STRIDE + kb]);
        afr.h[1] = *(const v8h*)(&lA[cur][mfrag * LDS_STRIDE + kb + 16]);

        v16h bf[4];
        #pragma unroll
        for (int j = 0; j < 4; ++j) {
            const int nf = 16 * j + (lane & 15);
            union { v16h v; v8h h[2]; } b;
            b.h[0] = *(const v8h*)(&lB[cur][nf * LDS_STRIDE + kb]);
            b.h[1] = *(const v8h*)(&lB[cur][nf * LDS_STRIDE + kb + 16]);
            bf[j] = b.v;
        }
        #pragma unroll
        for (int j = 0; j < 4; ++j)
            acc[j] = __builtin_amdgcn_wmma_f32_16x16x32_f16(
                         false, afr.v, false, bf[j], (short)0, acc[j], false, false);
    }

    // epilogue: C/D layout -> element (M = r + 8*(lane>>4), N = lane&15) per VGPR r
    const int mbase = blockM + 16 * wave + 8 * (lane >> 4);
    #pragma unroll
    for (int j = 0; j < 4; ++j) {
        const int n = blockN + 16 * j + (lane & 15);
        const float b = bias[n];
        #pragma unroll
        for (int r = 0; r < 8; ++r) {
            const int m = mbase + r;
            const float val = acc[j][r] + b;
            if (epilogue == 1) {
                const float g = 1.0f / (1.0f + __expf(-val));
                Ch[(size_t)m * N + n] = (_Float16)(mul[(size_t)m * N + n] * g);
            } else {
                Cf[(size_t)m * N + n] = val;
            }
        }
    }
}

// f32 -> f16 pack, float4 at a time
__global__ __launch_bounds__(256)
void cvt_f32_f16(const float* __restrict__ src, _Float16* __restrict__ dst, int n4)
{
    const int i = blockIdx.x * blockDim.x + threadIdx.x;
    if (i < n4) {
        float4 f = ((const float4*)src)[i];
        v4h h; h.x = (_Float16)f.x; h.y = (_Float16)f.y;
               h.z = (_Float16)f.z; h.w = (_Float16)f.w;
        ((v4h*)dst)[i] = h;
    }
}

// Online-softmax local attention: 1 thread per (query n, head h).
// qkv layout: [N, 3072] with q at col h*64, k at 1024+h*64, v at 2048+h*64.
__global__ __launch_bounds__(256)
void attn_kernel(const float* __restrict__ qkv, const float* __restrict__ pos_bias,
                 const int* __restrict__ offsets, float* __restrict__ out,
                 int Nq, int D3, int NO)
{
    const int n = blockIdx.x * blockDim.x + threadIdx.x;
    const int h = blockIdx.y;
    const float scale = 0.125f;          // 64^-0.5

    const float4* qp = (const float4*)(qkv + (size_t)n * D3 + h * 64);
    float4 q[16], acc[16];
    #pragma unroll
    for (int i = 0; i < 16; ++i) { q[i] = qp[i]; acc[i] = make_float4(0.f, 0.f, 0.f, 0.f); }

    float mrun = -INFINITY, l = 0.f;
    for (int o = 0; o < NO; ++o) {
        const int key = n - offsets[o];
        if (key < 0) continue;           // offset 0 always valid -> l > 0
        const float4* kp = (const float4*)(qkv + (size_t)key * D3 + 1024 + h * 64);
        float s = 0.f;
        #pragma unroll
        for (int i = 0; i < 16; ++i) {
            float4 kv = kp[i];
            s += q[i].x * kv.x + q[i].y * kv.y + q[i].z * kv.z + q[i].w * kv.w;
        }
        s = s * scale + pos_bias[o * 16 + h];
        const float mn = fmaxf(mrun, s);
        const float c  = __expf(mrun - mn);   // exp(-inf) = 0 on first hit
        const float p  = __expf(s - mn);
        l = l * c + p;
        const float4* vp = (const float4*)(qkv + (size_t)key * D3 + 2048 + h * 64);
        #pragma unroll
        for (int i = 0; i < 16; ++i) {
            float4 vv = vp[i];
            acc[i].x = acc[i].x * c + p * vv.x;
            acc[i].y = acc[i].y * c + p * vv.y;
            acc[i].z = acc[i].z * c + p * vv.z;
            acc[i].w = acc[i].w * c + p * vv.w;
        }
        mrun = mn;
    }

    const float inv = 1.0f / l;
    float4* op = (float4*)(out + (size_t)n * 1024 + h * 64);
    #pragma unroll
    for (int i = 0; i < 16; ++i)
        op[i] = make_float4(acc[i].x * inv, acc[i].y * inv, acc[i].z * inv, acc[i].w * inv);
}

extern "C" void kernel_launch(void* const* d_in, const int* in_sizes, int n_in,
                              void* d_out, int out_size, void* d_ws, size_t ws_size,
                              hipStream_t stream)
{
    const float* x     = (const float*)d_in[0];
    const float* Wqkv  = (const float*)d_in[1];
    const float* bqkv  = (const float*)d_in[2];
    const float* Wgate = (const float*)d_in[3];
    const float* bgate = (const float*)d_in[4];
    const float* Wout  = (const float*)d_in[5];
    const float* bout  = (const float*)d_in[6];
    const float* pbias = (const float*)d_in[7];
    const int*   offs  = (const int*)d_in[8];
    const int NO = in_sizes[8];          // 74 offsets

    const int M = 2048, D = 1024, D3 = 3072;

    // workspace carve-up (all 256B aligned by construction)
    char* w = (char*)d_ws;
    _Float16* xh     = (_Float16*)w;  w += (size_t)M  * D  * 2;   //  4.0 MB
    _Float16* Wqkvh  = (_Float16*)w;  w += (size_t)D3 * D  * 2;   //  6.3 MB
    _Float16* Wgateh = (_Float16*)w;  w += (size_t)D  * D  * 2;   //  2.1 MB
    _Float16* Wouth  = (_Float16*)w;  w += (size_t)D  * D  * 2;   //  2.1 MB
    float*    qkv    = (float*)w;     w += (size_t)M  * D3 * 4;   // 25.2 MB
    float*    attn   = (float*)w;     w += (size_t)M  * D  * 4;   //  8.4 MB
    _Float16* gatedh = (_Float16*)w;                              //  4.2 MB

    float* outp = (float*)d_out;
    dim3 blk(256);

    // 0) one-time f16 conversions (feeds the TDM byte-mover path)
    cvt_f32_f16<<<dim3((M * D / 4 + 255) / 256),  blk, 0, stream>>>(x,     xh,     M * D / 4);
    cvt_f32_f16<<<dim3((D3 * D / 4 + 255) / 256), blk, 0, stream>>>(Wqkv,  Wqkvh,  D3 * D / 4);
    cvt_f32_f16<<<dim3((D * D / 4 + 255) / 256),  blk, 0, stream>>>(Wgate, Wgateh, D * D / 4);
    cvt_f32_f16<<<dim3((D * D / 4 + 255) / 256),  blk, 0, stream>>>(Wout,  Wouth,  D * D / 4);

    // 1) qkv = x @ Wqkv^T + bqkv   (f32 out for the attention stage)
    gemm_wmma_f16<<<dim3(D3 / TILE_N, M / TILE_M), blk, 0, stream>>>(
        xh, Wqkvh, bqkv, nullptr, qkv, nullptr, M, D3, D, 0);

    // 2) sparse local attention -> attn[M, D] (f32)
    attn_kernel<<<dim3(M / 256, 16), blk, 0, stream>>>(qkv, pbias, offs, attn, M, D3, NO);

    // 3) gated(f16) = attn * sigmoid(x @ Wgate^T + bgate)
    gemm_wmma_f16<<<dim3(D / TILE_N, M / TILE_M), blk, 0, stream>>>(
        xh, Wgateh, bgate, attn, nullptr, gatedh, M, D, D, 1);

    // 4) out = gated @ Wout^T + bout  (f32 to d_out)
    gemm_wmma_f16<<<dim3(D / TILE_N, M / TILE_M), blk, 0, stream>>>(
        gatedh, Wouth, bout, nullptr, outp, nullptr, M, D, D, 0);
}
